// ActionModel_74397423501380
// MI455X (gfx1250) — matrix operand order repaired and checked by reading, not compiled, not run on hardware
//
#include <hip/hip_runtime.h>
#include <hip/hip_bf16.h>
#include <math.h>

typedef _Float16 v16h __attribute__((ext_vector_type(16)));
typedef _Float16 v8h  __attribute__((ext_vector_type(8)));
typedef float    v8f  __attribute__((ext_vector_type(8)));
typedef float    v4f  __attribute__((ext_vector_type(4)));
typedef int      v4i  __attribute__((ext_vector_type(4)));

#define N_NODES 65536
#define N_GRAPHS 64
#define NUM_E   1048576
#define F_NODE  128
#define F_EDGE  32
#define HDIM    128
#define ADIM    32
#define BN_EPS  1e-5f

// ---------------------------------------------------------------------------
// Prep: f16 transposed weights + folded BN affine (scale/shift)
// ---------------------------------------------------------------------------
__global__ __launch_bounds__(256) void prep_kernel(
    const float* __restrict__ We1, const float* __restrict__ We2,
    const float* __restrict__ W1,  const float* __restrict__ W2,
    const float* __restrict__ b1, const float* __restrict__ g1,
    const float* __restrict__ bt1, const float* __restrict__ m1, const float* __restrict__ v1,
    const float* __restrict__ b2, const float* __restrict__ g2,
    const float* __restrict__ bt2, const float* __restrict__ m2, const float* __restrict__ v2,
    _Float16* __restrict__ Wet1, _Float16* __restrict__ Wet2,
    _Float16* __restrict__ Wt1,  _Float16* __restrict__ Wt2,
    float* __restrict__ sc1, float* __restrict__ sh1,
    float* __restrict__ sc2, float* __restrict__ sh2)
{
  int i = blockIdx.x * 256 + threadIdx.x;
  if (i < F_NODE * F_EDGE) {                 // edge weights, transposed [n][k]
    int n = i / F_EDGE, k = i % F_EDGE;
    Wet1[i] = (_Float16)We1[k * F_NODE + n];
    Wet2[i] = (_Float16)We2[k * HDIM + n];
  }
  if (i < HDIM * HDIM) {                     // node weights, transposed [n][k]
    int n = i / HDIM, k = i % HDIM;
    Wt1[i] = (_Float16)W1[k * HDIM + n];
    Wt2[i] = (_Float16)W2[k * HDIM + n];
  }
  if (i < HDIM) {                            // fold bias+BN: out = sc*y + sh
    float s1 = g1[i] * rsqrtf(v1[i] + BN_EPS);
    sc1[i] = s1; sh1[i] = s1 * (b1[i] - m1[i]) + bt1[i];
    float s2 = g2[i] * rsqrtf(v2[i] + BN_EPS);
    sc2[i] = s2; sh2[i] = s2 * (b2[i] - m2[i]) + bt2[i];
  }
}

__global__ __launch_bounds__(256) void zero_kernel(float* __restrict__ p, int n4) {
  int i = blockIdx.x * 256 + threadIdx.x;
  if (i < n4) ((v4f*)p)[i] = (v4f){0.f, 0.f, 0.f, 0.f};
}

// ---------------------------------------------------------------------------
// GINE edge kernel: per wave32, one 16-edge tile.
//   msg = relu(x[src] + edge_attr @ We + be); agg[dst] += msg (f32 atomics)
// edge_attr tile [16x32] is exactly one K=32 f16 WMMA A operand.
// All addressing uses 32-bit element offsets (saddr + voffset + imm form);
// nloc is folded into the offset once, t*16 folds into the immediate.
// ---------------------------------------------------------------------------
__global__ __launch_bounds__(256) void gine_edge_kernel(
    const float* __restrict__ xsrc,       // [N,128] node features
    const int*   __restrict__ srcIdx,     // [E]
    const int*   __restrict__ dstIdx,     // [E]
    const float* __restrict__ ea,         // [E,32]
    const _Float16* __restrict__ WetH,    // [128(n) x 32(k)] f16
    const float* __restrict__ be,         // [128]
    float* __restrict__ agg)              // [N,128], pre-zeroed
{
  const int lane   = threadIdx.x & 31;
  const int laneHi = lane >> 4;
  const int nloc   = lane & 15;
  const int tile   = (blockIdx.x * blockDim.x + threadIdx.x) >> 5;
  const int ebase  = tile * 16;

  // B operands: We [32x16] per N-tile, resident in VGPRs.
  v16h B[8];
  {
    const int kb = laneHi ? 8 : 0;
#pragma unroll
    for (int t = 0; t < 8; ++t) {
      const _Float16* p = WetH + (t * 16 + nloc) * F_EDGE + kb;
      v8h lo = *(const v8h*)p;
      v8h hi = *(const v8h*)(p + 16);
      B[t] = __builtin_shufflevector(lo, hi, 0,1,2,3,4,5,6,7,8,9,10,11,12,13,14,15);
    }
  }
  float beArr[8];
#pragma unroll
  for (int t = 0; t < 8; ++t) beArr[t] = be[t * 16 + nloc];

  // A operand: 16 edges x 32 feats (ISA f16 A layout: runs {kA..+7},{kA+16..+23})
  const int eoff = (ebase + nloc) * F_EDGE + (laneHi ? 8 : 0);
  v4f e0 = *(const v4f*)(ea + eoff);
  v4f e1 = *(const v4f*)(ea + eoff + 4);
  v4f e2 = *(const v4f*)(ea + eoff + 16);
  v4f e3 = *(const v4f*)(ea + eoff + 20);
  v16h Aop;
#pragma unroll
  for (int i = 0; i < 4; ++i) {
    Aop[i]      = (_Float16)e0[i];
    Aop[i + 4]  = (_Float16)e1[i];
    Aop[i + 8]  = (_Float16)e2[i];
    Aop[i + 12] = (_Float16)e3[i];
  }

  // Edge src/dst indices for the 8 rows this half-wave owns (C layout: m = v + 8*laneHi)
  int sOff[8], dOff[8];
  {
    const int ib = ebase + 8 * laneHi;
    v4i s0 = *(const v4i*)(srcIdx + ib);
    v4i s1 = *(const v4i*)(srcIdx + ib + 4);
    v4i d0 = *(const v4i*)(dstIdx + ib);
    v4i d1 = *(const v4i*)(dstIdx + ib + 4);
#pragma unroll
    for (int i = 0; i < 4; ++i) {
      sOff[i]     = s0[i] * F_NODE + nloc;
      sOff[i + 4] = s1[i] * F_NODE + nloc;
      dOff[i]     = d0[i] * F_NODE + nloc;
      dOff[i + 4] = d1[i] * F_NODE + nloc;
    }
  }

  // 8 WMMAs -> msg tile [16 x 128] in f32 accumulators
  v8f acc[8];
#pragma unroll
  for (int t = 0; t < 8; ++t) {
    v8f cz = (v8f){0.f,0.f,0.f,0.f,0.f,0.f,0.f,0.f};
    acc[t] = __builtin_amdgcn_wmma_f32_16x16x32_f16(false, Aop, false, B[t],
                                                    (short)0, cz, false, false);
  }

  // Epilogue: per N-tile, batch 8 gathers, then 8 scatter-adds (L2-resident agg)
#pragma unroll
  for (int t = 0; t < 8; ++t) {
    float xg[8];
#pragma unroll
    for (int v = 0; v < 8; ++v) xg[v] = xsrc[sOff[v] + t * 16];
    const float bias = beArr[t];
#pragma unroll
    for (int v = 0; v < 8; ++v) {
      float c = fmaxf(acc[t][v] + xg[v] + bias, 0.0f);
      __hip_atomic_fetch_add(agg + dOff[v] + t * 16, c,
                             __ATOMIC_RELAXED, __HIP_MEMORY_SCOPE_AGENT);
    }
  }
}

// ---------------------------------------------------------------------------
// GINE node kernel: h = act(scale * ((x+agg) @ W) + shift)
// Per wave32: 16 rows x 128 cols, 4 K-chunks x 8 N-tiles of f16 WMMA.
// May run in place (hout == xin): each wave reads its rows before storing.
// ---------------------------------------------------------------------------
__global__ __launch_bounds__(256) void gine_node_kernel(
    const float* xin,                    // [N,128] (may alias hout)
    const float* __restrict__ agg,       // [N,128]
    const _Float16* __restrict__ WtH,    // [128(n) x 128(k)] f16
    const float* __restrict__ scale,
    const float* __restrict__ shift,
    float* hout,                         // [N,128]
    int applySigmoid)
{
  const int lane   = threadIdx.x & 31;
  const int laneHi = lane >> 4;
  const int nloc   = lane & 15;
  const int tile   = (blockIdx.x * blockDim.x + threadIdx.x) >> 5;
  const int rbase  = tile * 16;

  float scArr[8], shArr[8];
#pragma unroll
  for (int t = 0; t < 8; ++t) {
    scArr[t] = scale[t * 16 + nloc];
    shArr[t] = shift[t * 16 + nloc];
  }

  v8f acc[8];
#pragma unroll
  for (int t = 0; t < 8; ++t) acc[t] = (v8f){0.f,0.f,0.f,0.f,0.f,0.f,0.f,0.f};

  const int rowA = (rbase + nloc) * HDIM + (laneHi ? 8 : 0);  // A base (elements)
  const int kbB  = laneHi ? 8 : 0;
  const int bB   = nloc * HDIM + kbB;                          // B base (elements)

#pragma unroll
  for (int kk = 0; kk < 4; ++kk) {
    // A: z = x + agg (on the fly f32->f16), K-chunk of 32
    const int kA = rowA + kk * 32;
    v4f x0 = *(const v4f*)(xin + kA);
    v4f x1 = *(const v4f*)(xin + kA + 4);
    v4f x2 = *(const v4f*)(xin + kA + 16);
    v4f x3 = *(const v4f*)(xin + kA + 20);
    v4f g0 = *(const v4f*)(agg + kA);
    v4f g1 = *(const v4f*)(agg + kA + 4);
    v4f g2 = *(const v4f*)(agg + kA + 16);
    v4f g3 = *(const v4f*)(agg + kA + 20);
    v16h Aop;
#pragma unroll
    for (int i = 0; i < 4; ++i) {
      Aop[i]      = (_Float16)(x0[i] + g0[i]);
      Aop[i + 4]  = (_Float16)(x1[i] + g1[i]);
      Aop[i + 8]  = (_Float16)(x2[i] + g2[i]);
      Aop[i + 12] = (_Float16)(x3[i] + g3[i]);
    }
#pragma unroll
    for (int t = 0; t < 8; ++t) {
      const _Float16* p = WtH + bB + t * 16 * HDIM + kk * 32;
      v8h lo = *(const v8h*)p;
      v8h hi = *(const v8h*)(p + 16);
      v16h Bop = __builtin_shufflevector(lo, hi, 0,1,2,3,4,5,6,7,8,9,10,11,12,13,14,15);
      acc[t] = __builtin_amdgcn_wmma_f32_16x16x32_f16(false, Aop, false, Bop,
                                                      (short)0, acc[t], false, false);
    }
  }

  // Epilogue: folded BN + ReLU (+ sigmoid for conv2), store (saddr+imm form)
  int outOff[8];
#pragma unroll
  for (int v = 0; v < 8; ++v)
    outOff[v] = (rbase + v + 8 * laneHi) * HDIM + nloc;
#pragma unroll
  for (int t = 0; t < 8; ++t) {
#pragma unroll
    for (int v = 0; v < 8; ++v) {
      float c = fmaxf(acc[t][v] * scArr[t] + shArr[t], 0.0f);
      if (applySigmoid) c = 1.0f / (1.0f + __expf(-c));
      hout[outOff[v] + t * 16] = c;
    }
  }
}

// ---------------------------------------------------------------------------
// Mean pool over contiguous equal-size graphs (1024 nodes each)
// ---------------------------------------------------------------------------
__global__ __launch_bounds__(128) void pool_kernel(const float* __restrict__ h,
                                                   float* __restrict__ pooled) {
  const int g = blockIdx.x;        // graph
  const int f = threadIdx.x;       // feature
  const int npg = N_NODES / N_GRAPHS;
  float s = 0.f;
  const float* base = h + g * npg * HDIM + f;
  for (int i = 0; i < npg; ++i) s += base[i * HDIM];
  pooled[g * HDIM + f] = s * (1.0f / (float)npg);
}

// ---------------------------------------------------------------------------
// Head: Linear + BN + ReLU (64x128 @ 128x128); tiny, scalar is fine
// ---------------------------------------------------------------------------
__global__ __launch_bounds__(256) void head_fc_bn_relu(
    const float* __restrict__ in, const float* __restrict__ W,
    const float* __restrict__ b, const float* __restrict__ g,
    const float* __restrict__ bt, const float* __restrict__ m,
    const float* __restrict__ v, float* __restrict__ out) {
  int i = blockIdx.x * 256 + threadIdx.x;
  if (i >= N_GRAPHS * HDIM) return;
  int r = i / HDIM, n = i % HDIM;
  float acc = b[n];
  for (int k = 0; k < HDIM; ++k) acc += in[r * HDIM + k] * W[k * HDIM + n];
  float sc = g[n] * rsqrtf(v[n] + BN_EPS);
  out[i] = fmaxf(sc * (acc - m[n]) + bt[n], 0.0f);
}

__global__ __launch_bounds__(256) void head_out_kernel(
    const float* __restrict__ in, const float* __restrict__ W,
    const float* __restrict__ b, float* __restrict__ out) {
  int i = blockIdx.x * 256 + threadIdx.x;
  if (i >= N_GRAPHS * ADIM) return;
  int r = i / ADIM, n = i % ADIM;
  float acc = b[n];
  for (int k = 0; k < HDIM; ++k) acc += in[r * HDIM + k] * W[k * ADIM + n];
  out[i] = 1.0f / (1.0f + __expf(-acc));
}

// ---------------------------------------------------------------------------
extern "C" void kernel_launch(void* const* d_in, const int* in_sizes, int n_in,
                              void* d_out, int out_size, void* d_ws, size_t ws_size,
                              hipStream_t stream) {
  const float* x   = (const float*)d_in[0];
  const int*   ei  = (const int*)d_in[1];
  const float* ea  = (const float*)d_in[2];
  // d_in[3] = batch (unused: graphs are contiguous & equal size)
  const float* We1 = (const float*)d_in[4];  const float* be1 = (const float*)d_in[5];
  const float* W1  = (const float*)d_in[6];  const float* b1  = (const float*)d_in[7];
  const float* g1  = (const float*)d_in[8];  const float* bt1 = (const float*)d_in[9];
  const float* m1  = (const float*)d_in[10]; const float* v1  = (const float*)d_in[11];
  const float* We2 = (const float*)d_in[12]; const float* be2 = (const float*)d_in[13];
  const float* W2  = (const float*)d_in[14]; const float* b2  = (const float*)d_in[15];
  const float* g2  = (const float*)d_in[16]; const float* bt2 = (const float*)d_in[17];
  const float* m2  = (const float*)d_in[18]; const float* v2  = (const float*)d_in[19];
  const float* Wa1 = (const float*)d_in[20]; const float* ba1 = (const float*)d_in[21];
  const float* ga1 = (const float*)d_in[22]; const float* bta1= (const float*)d_in[23];
  const float* ma1 = (const float*)d_in[24]; const float* va1 = (const float*)d_in[25];
  const float* Wa2 = (const float*)d_in[26]; const float* ba2 = (const float*)d_in[27];
  const float* ga2 = (const float*)d_in[28]; const float* bta2= (const float*)d_in[29];
  const float* ma2 = (const float*)d_in[30]; const float* va2 = (const float*)d_in[31];
  const float* Wa3 = (const float*)d_in[32]; const float* ba3 = (const float*)d_in[33];

  const int* srcIdx = ei;
  const int* dstIdx = ei + NUM_E;

  // workspace carve-up (all 256B aligned)
  char* ws = (char*)d_ws;
  size_t off = 0;
  auto carve = [&](size_t bytes) { char* p = ws + off; off = (off + bytes + 255) & ~(size_t)255; return p; };
  float*    agg   = (float*)   carve((size_t)N_NODES * HDIM * 4);
  float*    h     = (float*)   carve((size_t)N_NODES * HDIM * 4);
  _Float16* Wet1  = (_Float16*)carve((size_t)HDIM * F_EDGE * 2);
  _Float16* Wet2  = (_Float16*)carve((size_t)HDIM * F_EDGE * 2);
  _Float16* Wt1   = (_Float16*)carve((size_t)HDIM * HDIM * 2);
  _Float16* Wt2   = (_Float16*)carve((size_t)HDIM * HDIM * 2);
  float*    sc1   = (float*)   carve(HDIM * 4);
  float*    sh1   = (float*)   carve(HDIM * 4);
  float*    sc2   = (float*)   carve(HDIM * 4);
  float*    sh2   = (float*)   carve(HDIM * 4);
  float*    pooled= (float*)   carve((size_t)N_GRAPHS * HDIM * 4);
  float*    a1    = (float*)   carve((size_t)N_GRAPHS * HDIM * 4);
  float*    a2    = (float*)   carve((size_t)N_GRAPHS * HDIM * 4);

  const int zero4   = N_NODES * HDIM / 4;          // 2,097,152 float4s
  const int zeroBlk = (zero4 + 255) / 256;         // 8192
  const int edgeBlk = (NUM_E / 16) * 32 / 256;     // 8192 blocks (1 tile / wave)
  const int nodeBlk = (N_NODES / 16) * 32 / 256;   // 512 blocks

  // 0) prep: f16 transposed weights + folded BN
  prep_kernel<<<64, 256, 0, stream>>>(We1, We2, W1, W2,
      b1, g1, bt1, m1, v1, b2, g2, bt2, m2, v2,
      Wet1, Wet2, Wt1, Wt2, sc1, sh1, sc2, sh2);

  // 1) conv1
  zero_kernel<<<zeroBlk, 256, 0, stream>>>(agg, zero4);
  gine_edge_kernel<<<edgeBlk, 256, 0, stream>>>(x, srcIdx, dstIdx, ea, Wet1, be1, agg);
  gine_node_kernel<<<nodeBlk, 256, 0, stream>>>(x, agg, Wt1, sc1, sh1, h, 0);

  // 2) conv2 (node update in place: h -> h)
  zero_kernel<<<zeroBlk, 256, 0, stream>>>(agg, zero4);
  gine_edge_kernel<<<edgeBlk, 256, 0, stream>>>(h, srcIdx, dstIdx, ea, Wet2, be2, agg);
  gine_node_kernel<<<nodeBlk, 256, 0, stream>>>(h, agg, Wt2, sc2, sh2, h, 1);

  // 3) pool + head
  pool_kernel<<<N_GRAPHS, 128, 0, stream>>>(h, pooled);
  head_fc_bn_relu<<<32, 256, 0, stream>>>(pooled, Wa1, ba1, ga1, bta1, ma1, va1, a1);
  head_fc_bn_relu<<<32, 256, 0, stream>>>(a1, Wa2, ba2, ga2, bta2, ma2, va2, a2);
  head_out_kernel<<<8, 256, 0, stream>>>(a2, Wa3, ba3, (float*)d_out);
}
